// FlowPolicy_81329500717064
// MI455X (gfx1250) — matrix operand order repaired and checked by reading, not compiled, not run on hardware
//
#include <hip/hip_runtime.h>
#include <cstdint>
#include <cstddef>

typedef __attribute__((ext_vector_type(16))) __bf16 bf16x16;
typedef __attribute__((ext_vector_type(8)))  float  fx8;

union FragAB { unsigned int u[8]; bf16x16 v; };

static constexpr int Bn = 32768;
static constexpr int Dd = 16;
static constexpr int Hh = 256;
static constexpr int Cc = 128;

// workspace byte offsets (all 64B aligned)
static constexpr size_t OFF_W0M  = 0;         // 16 B-frags (256x32, M0 mask, K zero-padded)
static constexpr size_t OFF_WFM  = 16384;     // 16 B-frags (32x256, MF mask)
static constexpr size_t OFF_WC0  = 32768;     // 64 B-frags (256x128)
static constexpr size_t OFF_WC   = 98304;     // 2*64 frags
static constexpr size_t OFF_W1   = 229376;    // 2*128 frags (MH mask)
static constexpr size_t OFF_W2   = 491520;    // 2*128 frags (MH mask)
static constexpr size_t OFF_L    = 753664;    // 16x16 f32 (unit lower)
static constexpr size_t OFF_U    = 754688;    // 16x16 f32
static constexpr size_t OFF_INV  = 755712;    // 16 i32 inverse perm
static constexpr size_t OFF_SLD  = 755776;    // f32 sum(log diag U)
static constexpr size_t OFF_CTX0 = 756224;    // B*256 bf16
static constexpr size_t OFF_GATE = 17533440;  // 2*B*256 bf16 (sigmoid applied)

static __device__ __forceinline__ unsigned short f2bf(float f){
  unsigned int x = __float_as_uint(f);
  x += 0x7fffu + ((x >> 16) & 1u);
  return (unsigned short)(x >> 16);
}
static __device__ __forceinline__ float bf2f(unsigned short h){
  return __uint_as_float(((unsigned int)h) << 16);
}
static __device__ __forceinline__ int degh(int x){ return (x % 15) + 1; }

// fast hardware transcendentals: v_exp_f32 / v_log_f32 / v_rcp_f32
static __device__ __forceinline__ float fast_sigmoid(float x){
  float e = __builtin_amdgcn_exp2f(-1.44269504088896f * x);
  return __builtin_amdgcn_rcpf(1.0f + e);
}
static __device__ __forceinline__ float fast_log(float x){
  return 0.69314718055994531f * __builtin_amdgcn_logf(x);
}

// relu on two packed bf16 (sign-bit masking)
static __device__ __forceinline__ unsigned int relu2(unsigned int x){
  unsigned int lo = (x >> 15) & 1u;
  unsigned int hi = (x >> 31) & 1u;
  return x & ((0x0000FFFFu * (1u - lo)) | (0xFFFF0000u * (1u - hi)));
}

// Gather a 16x32 bf16 A-fragment from row-major LDS (contiguous k order: k = 16*g + s).
// Each lane's fragment = two contiguous 16B runs -> two ds_load_b128.
static __device__ __forceinline__ bf16x16 ldsA(const unsigned short* sh, int stride,
                                               int rows0, int kbase, int lane){
  int g = lane >> 4, m = lane & 15;
  const unsigned short* pr = sh + (rows0 + m) * stride + kbase + g * 16;
  uint4 a = *reinterpret_cast<const uint4*>(pr);        // K = g*16 + 0..7
  uint4 b = *reinterpret_cast<const uint4*>(pr + 8);    // K = g*16 + 8..15
  FragAB f;
  f.u[0]=a.x; f.u[1]=a.y; f.u[2]=a.z; f.u[3]=a.w;
  f.u[4]=b.x; f.u[5]=b.y; f.u[6]=b.z; f.u[7]=b.w;
  return f.v;
}

// Load a packed 32x16 bf16 B-fragment from global (lane-major, 16 contiguous per lane).
static __device__ __forceinline__ bf16x16 gB(const char* frag, int lane){
  const uint4* p = reinterpret_cast<const uint4*>(frag);
  uint4 q0 = p[lane * 2];
  uint4 q1 = p[lane * 2 + 1];
  FragAB f;
  f.u[0]=q0.x; f.u[1]=q0.y; f.u[2]=q0.z; f.u[3]=q0.w;
  f.u[4]=q1.x; f.u[5]=q1.y; f.u[6]=q1.z; f.u[7]=q1.w;
  return f.v;
}

// ---------------- prep: W0 frags, L/U build, inv perm, sum log diag ----------------
__global__ void k_prep_misc(const float* __restrict__ W0, const float* __restrict__ lower,
                            const float* __restrict__ upper, const float* __restrict__ u_diag,
                            const int* __restrict__ perm, char* __restrict__ ws)
{
  int t = threadIdx.x;
  // Pack (W0*M0)^T into 16 B-fragments with K padded 16->32 (zeros).
  unsigned short* w0f = (unsigned short*)(ws + OFF_W0M);
  for (int q = 0; q < 32; ++q){
    int e = q * 256 + t;
    int frag = e >> 9, r = e & 511;
    int lane = r >> 4, elem = r & 15;
    int n = frag * 16 + (lane & 15);
    int k = (lane >> 4) * 16 + elem;
    float v = (k < 16 && degh(n) >= k + 1) ? W0[n * Dd + k] : 0.0f;
    w0f[e] = f2bf(v);
  }
  float* Lm = (float*)(ws + OFF_L);
  float* Um = (float*)(ws + OFF_U);
  Lm[t] = ((t >> 4) == (t & 15)) ? 1.0f : 0.0f;
  Um[t] = 0.0f;
  __syncthreads();
  if (t < 120){
    int rem = t, i = 1;
    while (rem >= i){ rem -= i; ++i; }
    Lm[i * 16 + rem] = lower[t];
    int rem2 = t, i2 = 0;
    while (rem2 >= 15 - i2){ rem2 -= 15 - i2; ++i2; }
    Um[i2 * 16 + (i2 + 1 + rem2)] = upper[t];
  }
  if (t < 16){
    float x = u_diag[t];
    float sp = (x > 20.f) ? x : log1pf(expf(x));
    Um[t * 16 + t] = sp + 1e-3f;
    ((int*)(ws + OFF_INV))[perm[t]] = t;
  }
  __syncthreads();
  if (t == 0){
    float s = 0.f;
    for (int i = 0; i < 16; ++i) s += logf(Um[i * 16 + i]);
    *((float*)(ws + OFF_SLD)) = s;
  }
}

// ---------------- pack: mask + bf16 + WMMA-B fragment layout ----------------
// mode: 0=no mask, 1=MH (degh(n)>=degh(k)), 2=MF ((n/2)+1 > degh(k))
__global__ void k_pack(const float* __restrict__ src, unsigned short* __restrict__ dst,
                       int N, int K, int mode, int total)
{
  int KT = K >> 5;
  for (int e = blockIdx.x * blockDim.x + threadIdx.x; e < total; e += gridDim.x * blockDim.x){
    int frag = e >> 9, r = e & 511;
    int lane = r >> 4, elem = r & 15;
    int kt = frag % KT, nt = frag / KT;
    int n = nt * 16 + (lane & 15);
    int k = kt * 32 + (lane >> 4) * 16 + elem;
    float v = src[(size_t)n * K + k];
    if (mode == 1){ if (!(degh(n) >= degh(k))) v = 0.f; }
    else if (mode == 2){ if (!(((n >> 1) + 1) > degh(k))) v = 0.f; }
    dst[e] = f2bf(v);
  }
}

// ---------------- ctx: ctx0 = state@Wc0.T + bc0 ; gates = sigmoid(state@Wc.T + bc) ----
__global__ void __launch_bounds__(256)
k_ctx(const float* __restrict__ state, const float* __restrict__ bc0,
      const float* __restrict__ bc, char* __restrict__ ws)
{
  __shared__ __align__(16) unsigned short sh_s[64 * 128];
  int tid = threadIdx.x;
  int grow0 = blockIdx.x * 64;
  for (int idx = tid; idx < 64 * 128; idx += 256){
    int rr = idx >> 7, cc = idx & 127;
    sh_s[idx] = f2bf(state[(size_t)(grow0 + rr) * 128 + cc]);
  }
  __syncthreads();
  int w = tid >> 5, lane = tid & 31;
  int tile = w >> 1, half = w & 1;
  int rows0 = tile * 16;
  #pragma unroll 1
  for (int tgt = 0; tgt < 3; ++tgt){
    const char* fragBase = ws + ((tgt == 0) ? OFF_WC0 : OFF_WC + (size_t)(tgt - 1) * 65536);
    const float* bias = (tgt == 0) ? bc0 : (bc + (tgt - 1) * 256);
    unsigned short* dst = (unsigned short*)(ws + ((tgt == 0) ? OFF_CTX0
                                   : OFF_GATE + (size_t)(tgt - 1) * Bn * 256 * 2));
    #pragma unroll 1
    for (int n8 = 0; n8 < 8; ++n8){
      int nt = half * 8 + n8;
      bf16x16 bfr[4];
      #pragma unroll
      for (int kt = 0; kt < 4; ++kt)
        bfr[kt] = gB(fragBase + (size_t)(nt * 4 + kt) * 1024, lane);
      fx8 acc = (fx8)0.0f;
      #pragma unroll
      for (int kt = 0; kt < 4; ++kt){
        bf16x16 av = ldsA(sh_s, 128, rows0, kt * 32, lane);
        acc = __builtin_amdgcn_wmma_f32_16x16x32_bf16(false, av, false, bfr[kt],
                                                      (short)0, acc, false, false);
      }
      int colL = lane & 15, gg = lane >> 4;
      #pragma unroll
      for (int r = 0; r < 8; ++r){
        int row = rows0 + r + 8 * gg;
        int col = nt * 16 + colL;
        float v = acc[r] + bias[col];
        if (tgt > 0) v = fast_sigmoid(v);
        dst[(size_t)(grow0 + row) * 256 + col] = f2bf(v);
      }
    }
  }
}

// ---------------- main flow kernel: 128 rows / workgroup, 16 MADE iterations ---------
// 8 waves; wave w owns rows [w*16, w*16+16) and covers all 16 N-tiles in 2 passes.
__global__ void __launch_bounds__(256)
k_flow(const float* __restrict__ eps, const float* __restrict__ b0,
       const float* __restrict__ b1, const float* __restrict__ b2,
       const float* __restrict__ bfv, const float* __restrict__ lu_bias,
       char* __restrict__ ws, float* __restrict__ out)
{
  __shared__ __align__(16) unsigned short sh_h[128 * 256];  // 64 KB bf16 residual stream h
  __shared__ __align__(16) unsigned short sh_a[128 * 256];  // 64 KB relu(h)
  __shared__ __align__(16) unsigned short sh_t[128 * 256];  // 64 KB relu(t1)
  __shared__ __align__(16) float          sh_p[128 * 32];   // 16 KB final projection
  __shared__ __align__(16) float          sh_z[128 * 16];   // 8 KB  z (f32, for LU solve)
  __shared__ __align__(16) unsigned short sh_zb[128 * 32];  // 8 KB  z bf16, K zero-padded
  __shared__ __align__(16) float          sh_ls[128 * 16];  // 8 KB  last log(scale)
  __shared__ float sh_L[256];
  __shared__ float sh_U[256];

  int tid = threadIdx.x;
  int grow0 = blockIdx.x * 128;
  int w = tid >> 5, lane = tid & 31;
  int rows0 = w * 16;

  const char* wsW0 = ws + OFF_W0M;
  const char* wsWf = ws + OFF_WFM;
  const char* wsW1 = ws + OFF_W1;
  const char* wsW2 = ws + OFF_W2;
  const unsigned short* ctx0  = (const unsigned short*)(ws + OFF_CTX0);
  const unsigned short* gates = (const unsigned short*)(ws + OFF_GATE);
  const int* invp = (const int*)(ws + OFF_INV);

  sh_L[tid] = ((const float*)(ws + OFF_L))[tid];
  sh_U[tid] = ((const float*)(ws + OFF_U))[tid];
  for (int idx = tid; idx < 128 * 16; idx += 256) sh_z[idx] = 0.f;
  for (int idx = tid; idx < 128 * 32; idx += 256) sh_zb[idx] = 0;  // includes K-pad zeros
  __syncthreads();

  int colL = lane & 15, gg = lane >> 4;

  #pragma unroll 1
  for (int it = 0; it < 16; ++it){
    // ---- input layer (WMMA, K=32 zero-padded): h = z @ (W0*M0).T + b0 + ctx0 ----
    {
      bf16x16 zfrag = ldsA(sh_zb, 32, rows0, 0, lane);
      #pragma unroll 1
      for (int nt = 0; nt < 16; ++nt){
        bf16x16 bv = gB(wsW0 + (size_t)nt * 1024, lane);
        fx8 acc = (fx8)0.0f;
        acc = __builtin_amdgcn_wmma_f32_16x16x32_bf16(false, zfrag, false, bv,
                                                      (short)0, acc, false, false);
        #pragma unroll
        for (int r = 0; r < 8; ++r){
          int row = rows0 + r + 8 * gg;
          int col = nt * 16 + colL;
          float v = acc[r] + b0[col] + bf2f(ctx0[(size_t)(grow0 + row) * 256 + col]);
          sh_h[row * 256 + col] = f2bf(v);
        }
      }
    }
    __syncthreads();

    #pragma unroll 1
    for (int blk = 0; blk < 2; ++blk){
      // stage a = relu(h), pure bf16 sign masking on uint4
      #pragma unroll 4
      for (int q = 0; q < 32; ++q){
        int i = q * 256 + tid;
        uint4 x = reinterpret_cast<const uint4*>(sh_h)[i];
        x.x = relu2(x.x); x.y = relu2(x.y); x.z = relu2(x.z); x.w = relu2(x.w);
        reinterpret_cast<uint4*>(sh_a)[i] = x;
      }
      __syncthreads();

      // t1 = a @ (W1*MH).T + b1 ; relu -> sh_t   (2 passes x 8 N-tiles)
      {
        const char* wbase = wsW1 + (size_t)blk * 131072;
        #pragma unroll 1
        for (int pass = 0; pass < 2; ++pass){
          fx8 acc[8];
          #pragma unroll
          for (int n = 0; n < 8; ++n) acc[n] = (fx8)0.0f;
          #pragma unroll 1
          for (int kt = 0; kt < 8; ++kt){
            bf16x16 bfr[8];
            #pragma unroll
            for (int n = 0; n < 8; ++n)
              bfr[n] = gB(wbase + (size_t)((pass * 8 + n) * 8 + kt) * 1024, lane);
            if (kt < 7)
              __builtin_prefetch(wbase + (size_t)(pass * 64 + kt + 1) * 1024 + lane * 32, 0, 1);
            bf16x16 av = ldsA(sh_a, 256, rows0, kt * 32, lane);
            #pragma unroll
            for (int n = 0; n < 8; ++n)
              acc[n] = __builtin_amdgcn_wmma_f32_16x16x32_bf16(false, av, false, bfr[n],
                                                               (short)0, acc[n], false, false);
          }
          #pragma unroll
          for (int n = 0; n < 8; ++n){
            int col = (pass * 8 + n) * 16 + colL;
            float bb = b1[blk * 256 + col];
            #pragma unroll
            for (int r = 0; r < 8; ++r){
              int row = rows0 + r + 8 * gg;
              sh_t[row * 256 + col] = f2bf(fmaxf(acc[n][r] + bb, 0.f));
            }
          }
        }
      }
      __syncthreads();

      // t2 = t1 @ (W2*MH).T + b2 ; h += t2 * gate
      {
        const char* wbase = wsW2 + (size_t)blk * 131072;
        #pragma unroll 1
        for (int pass = 0; pass < 2; ++pass){
          fx8 acc[8];
          #pragma unroll
          for (int n = 0; n < 8; ++n) acc[n] = (fx8)0.0f;
          #pragma unroll 1
          for (int kt = 0; kt < 8; ++kt){
            bf16x16 bfr[8];
            #pragma unroll
            for (int n = 0; n < 8; ++n)
              bfr[n] = gB(wbase + (size_t)((pass * 8 + n) * 8 + kt) * 1024, lane);
            if (kt < 7)
              __builtin_prefetch(wbase + (size_t)(pass * 64 + kt + 1) * 1024 + lane * 32, 0, 1);
            bf16x16 av = ldsA(sh_t, 256, rows0, kt * 32, lane);
            #pragma unroll
            for (int n = 0; n < 8; ++n)
              acc[n] = __builtin_amdgcn_wmma_f32_16x16x32_bf16(false, av, false, bfr[n],
                                                               (short)0, acc[n], false, false);
          }
          #pragma unroll
          for (int n = 0; n < 8; ++n){
            int col = (pass * 8 + n) * 16 + colL;
            float bb = b2[blk * 256 + col];
            #pragma unroll
            for (int r = 0; r < 8; ++r){
              int row = rows0 + r + 8 * gg;
              float g = bf2f(gates[(size_t)blk * Bn * 256 + (size_t)(grow0 + row) * 256 + col]);
              float hv = bf2f(sh_h[row * 256 + col]);
              sh_h[row * 256 + col] = f2bf(hv + (acc[n][r] + bb) * g);
            }
          }
        }
      }
      __syncthreads();
    }

    // ---- final projection p = h @ (Wf*MF).T + bf  (A read directly from bf16 sh_h) ----
    {
      fx8 acc[2];
      acc[0] = (fx8)0.0f; acc[1] = (fx8)0.0f;
      #pragma unroll 1
      for (int kt = 0; kt < 8; ++kt){
        bf16x16 av = ldsA(sh_h, 256, rows0, kt * 32, lane);
        #pragma unroll
        for (int n = 0; n < 2; ++n){
          bf16x16 bv = gB(wsWf + (size_t)(n * 8 + kt) * 1024, lane);
          acc[n] = __builtin_amdgcn_wmma_f32_16x16x32_bf16(false, av, false, bv,
                                                           (short)0, acc[n], false, false);
        }
      }
      #pragma unroll
      for (int n = 0; n < 2; ++n){
        int col = n * 16 + colL;
        float bb = bfv[col];
        #pragma unroll
        for (int r = 0; r < 8; ++r){
          int row = rows0 + r + 8 * gg;
          sh_p[row * 32 + col] = acc[n][r] + bb;
        }
      }
    }
    __syncthreads();

    // ---- z = (eps - shift) * rcp(scale) ; scale = sigmoid(u+2)+1e-3 ----
    #pragma unroll
    for (int q = 0; q < 8; ++q){
      int pair = q * 256 + tid;
      int row = pair >> 4, dim = pair & 15;
      float u  = sh_p[row * 32 + 2 * dim];
      float sf = sh_p[row * 32 + 2 * dim + 1];
      float sc = fast_sigmoid(u + 2.0f) + 1e-3f;
      float e  = eps[(size_t)(grow0 + row) * 16 + dim];
      float zv = (e - sf) * __builtin_amdgcn_rcpf(sc);
      sh_z[row * 16 + dim]  = zv;
      sh_zb[row * 32 + dim] = f2bf(zv);
      sh_ls[row * 16 + dim] = fast_log(sc);
    }
    __syncthreads();
  }

  // ---- epilogue: LU solve, inverse permutation, tanh squash, log prob ----
  if (tid < 128){
    int row = tid; int grow = grow0 + row;
    float zr[16], wv[16], zz[16];
    float lq = -8.0f * logf(6.283185307179586f);
    lq += *((const float*)(ws + OFF_SLD));
    #pragma unroll
    for (int d = 0; d < 16; ++d){
      float e = eps[(size_t)grow * 16 + d];
      lq += sh_ls[row * 16 + d] - 0.5f * e * e;
      zr[d] = sh_z[row * 16 + d] - lu_bias[d];
    }
    #pragma unroll 1
    for (int i = 0; i < 16; ++i){
      float s = zr[i];
      for (int j = 0; j < i; ++j) s -= sh_L[i * 16 + j] * wv[j];
      wv[i] = s;
    }
    #pragma unroll 1
    for (int i = 15; i >= 0; --i){
      float s = wv[i];
      for (int j = i + 1; j < 16; ++j) s -= sh_U[i * 16 + j] * zz[j];
      zz[i] = s / sh_U[i * 16 + i];
    }
    #pragma unroll
    for (int j = 0; j < 16; ++j){
      float zv = zz[invp[j]];
      float a = tanhf(zv);
      out[(size_t)grow * 16 + j] = a;                         // actions
      out[(size_t)Bn * 17 + (size_t)grow * 16 + j] = zv;      // mean_action
      lq -= fast_log(1.0f - a * a + 1e-6f);
    }
    out[(size_t)Bn * 16 + grow] = lq;                         // log_probs
  }
}

extern "C" void kernel_launch(void* const* d_in, const int* in_sizes, int n_in,
                              void* d_out, int out_size, void* d_ws, size_t ws_size,
                              hipStream_t stream)
{
  const float* state  = (const float*)d_in[0];
  const float* eps    = (const float*)d_in[1];
  const float* W0     = (const float*)d_in[2];
  const float* b0     = (const float*)d_in[3];
  const float* Wc0    = (const float*)d_in[4];
  const float* bc0    = (const float*)d_in[5];
  const float* W1     = (const float*)d_in[6];
  const float* b1     = (const float*)d_in[7];
  const float* W2     = (const float*)d_in[8];
  const float* b2     = (const float*)d_in[9];
  const float* Wc     = (const float*)d_in[10];
  const float* bc     = (const float*)d_in[11];
  const float* Wf     = (const float*)d_in[12];
  const float* bfv    = (const float*)d_in[13];
  const float* lower  = (const float*)d_in[14];
  const float* upper  = (const float*)d_in[15];
  const float* u_diag = (const float*)d_in[16];
  const float* lu_b   = (const float*)d_in[17];
  const int*   perm   = (const int*)d_in[18];
  char* ws = (char*)d_ws;
  float* out = (float*)d_out;
  (void)in_sizes; (void)n_in; (void)out_size; (void)ws_size;

  k_prep_misc<<<1, 256, 0, stream>>>(W0, lower, upper, u_diag, perm, ws);
  k_pack<<<128, 256, 0, stream>>>(Wc0, (unsigned short*)(ws + OFF_WC0), 256, 128, 0, 32768);
  for (int i = 0; i < 2; ++i)
    k_pack<<<128, 256, 0, stream>>>(Wc + (size_t)i * Hh * Cc,
                                    (unsigned short*)(ws + OFF_WC + (size_t)i * 65536),
                                    256, 128, 0, 32768);
  for (int i = 0; i < 2; ++i)
    k_pack<<<256, 256, 0, stream>>>(W1 + (size_t)i * Hh * Hh,
                                    (unsigned short*)(ws + OFF_W1 + (size_t)i * 131072),
                                    256, 256, 1, 65536);
  for (int i = 0; i < 2; ++i)
    k_pack<<<256, 256, 0, stream>>>(W2 + (size_t)i * Hh * Hh,
                                    (unsigned short*)(ws + OFF_W2 + (size_t)i * 131072),
                                    256, 256, 1, 65536);
  k_pack<<<32, 256, 0, stream>>>(Wf, (unsigned short*)(ws + OFF_WFM), 32, 256, 2, 8192);

  k_ctx<<<Bn / 64, 256, 0, stream>>>(state, bc0, bc, ws);
  k_flow<<<Bn / 128, 256, 0, stream>>>(eps, b0, b1, b2, bfv, lu_b, ws, out);
}
